// feedforward_17222818857346
// MI455X (gfx1250) — matrix operand order repaired and checked by reading, not compiled, not run on hardware
//
#include <hip/hip_runtime.h>

typedef __attribute__((ext_vector_type(16))) _Float16 v16h;
typedef __attribute__((ext_vector_type(8)))  float    v8f;

#if __has_builtin(__builtin_amdgcn_tanhf)
__device__ __forceinline__ float fast_tanh(float x) { return __builtin_amdgcn_tanhf(x); }
#else
__device__ __forceinline__ float fast_tanh(float x) { return tanhf(x); }
#endif

// ---------------------------------------------------------------------------
// A-matrix builder (per-layer weights, loaded once per wave).
// 16-bit A 16x32 layout: lanes 0-15, element e  -> (M=lane,   K=e)
//                        lanes 16-31, element e -> (M=lane-16, K=8+e)  [e<8]
//                        elements 8..15         -> K=16..31 (0 here except the bias at e=8, lanes 0-15)
// We place W[m][k] at K=k (k<IN) and the bias b[m] at K=16 (lanes 0-15, e=8).
// B supplies 1.0 at K=16, so the WMMA computes  D = W * x + b  directly.
// ---------------------------------------------------------------------------
template <int OUT, int IN>
__device__ __forceinline__ v16h make_A(const float* __restrict__ W,
                                       const float* __restrict__ b,
                                       int lane) {
  v16h a;
#pragma unroll
  for (int e = 0; e < 16; ++e) a[e] = (_Float16)0.0f;
  const int m = lane & 15;
  if (lane < 16) {
    if (m < OUT) {
#pragma unroll
      for (int e = 0; e < 8; ++e)
        if (e < IN) a[e] = (_Float16)W[m * IN + e];
      a[8] = (_Float16)b[m];  // K=16 bias column
    }
  }
  return a;
}

// ---------------------------------------------------------------------------
// B-matrix from f32 activations in D layout (lane = batch column, t[r] = feat r).
// 16-bit B 32x16 layout: lanes 0-15, element e -> (K=e for e<8, K=8+e for e>=8, N=lane);
// element 8 (K=16) carries the constant 1.0 that multiplies the bias row of A.
// Lanes 16-31 map to K=8..15,24..31 which hit all-zero A entries -> don't care.
// ---------------------------------------------------------------------------
__device__ __forceinline__ v16h pack_B(const float* __restrict__ t) {
  v16h b;
#pragma unroll
  for (int e = 0; e < 8; ++e) b[e] = (_Float16)t[e];
  b[8] = (_Float16)1.0f;  // bias multiplier at K=16
#pragma unroll
  for (int e = 9; e < 16; ++e) b[e] = (_Float16)0.0f;
  return b;
}

// One layer: D = A*B (+bias via K=16), optional f32 residual, tanh, f32 out.
template <int OUT, bool RES>
__device__ __forceinline__ void wmma_layer(const v16h& A, const v16h& Bm,
                                           float* __restrict__ t,
                                           const float* __restrict__ res) {
  v8f c = {};
  v8f d = __builtin_amdgcn_wmma_f32_16x16x32_f16(
      false, A, false, Bm, (short)0, c, false, false);
#pragma unroll
  for (int r = 0; r < 8; ++r) {
    if (r < OUT) {
      float v = d[r];
      if (RES) v += res[r];
      t[r] = fast_tanh(v);
    } else {
      t[r] = 0.0f;
    }
  }
}

__global__ __launch_bounds__(256) void mlp12_wmma_kernel(
    const float* __restrict__ x, float* __restrict__ out, int ntiles,
    const float* __restrict__ W1,  const float* __restrict__ b1,
    const float* __restrict__ W2,  const float* __restrict__ b2,
    const float* __restrict__ W3,  const float* __restrict__ b3,
    const float* __restrict__ W4,  const float* __restrict__ b4,
    const float* __restrict__ W5,  const float* __restrict__ b5,
    const float* __restrict__ W6,  const float* __restrict__ b6,
    const float* __restrict__ W7,  const float* __restrict__ b7,
    const float* __restrict__ W8,  const float* __restrict__ b8,
    const float* __restrict__ W9,  const float* __restrict__ b9,
    const float* __restrict__ W10, const float* __restrict__ b10,
    const float* __restrict__ W11, const float* __restrict__ b11,
    const float* __restrict__ W12, const float* __restrict__ b12) {
  const int lane   = threadIdx.x & 31;
  const int m      = lane & 15;
  const bool lo    = lane < 16;
  const int wave   = blockIdx.x * (blockDim.x >> 5) + (threadIdx.x >> 5);
  const int nwaves = gridDim.x * (blockDim.x >> 5);

  // Hoisted per-wave weight/bias setup (A matrices carry bias in the K=16 slot).
  const v16h A1  = make_A<8, 8>(W1,  b1,  lane);
  const v16h A2  = make_A<8, 8>(W2,  b2,  lane);
  const v16h A3  = make_A<8, 8>(W3,  b3,  lane);
  const v16h A4  = make_A<8, 8>(W4,  b4,  lane);
  const v16h A5  = make_A<8, 8>(W5,  b5,  lane);
  const v16h A6  = make_A<8, 8>(W6,  b6,  lane);
  const v16h A7  = make_A<8, 8>(W7,  b7,  lane);
  const v16h A8  = make_A<8, 8>(W8,  b8,  lane);
  const v16h A9  = make_A<8, 8>(W9,  b9,  lane);
  const v16h A10 = make_A<5, 8>(W10, b10, lane);
  const v16h A11 = make_A<3, 5>(W11, b11, lane);
  const v16h A12 = make_A<1, 3>(W12, b12, lane);

  for (int tile = wave; tile < ntiles; tile += nwaves) {
    // Load 16 rows of x: lane m (m<16) owns row tile*16+m (8 floats = 32B).
    float xr[8];
#pragma unroll
    for (int e = 0; e < 8; ++e) xr[e] = 0.0f;
    if (lo) {
      const float4* p =
          reinterpret_cast<const float4*>(x + (size_t)(tile * 16 + m) * 8);
      const float4 a0 = p[0];
      const float4 a1 = p[1];
      xr[0] = a0.x; xr[1] = a0.y; xr[2] = a0.z; xr[3] = a0.w;
      xr[4] = a1.x; xr[5] = a1.y; xr[6] = a1.z; xr[7] = a1.w;
    }
    v16h Bv = pack_B(xr);

    float t[8], t2[8], t5[8], t7[8];
    wmma_layer<8, false>(A1, Bv, t,  nullptr); Bv = pack_B(t);
    wmma_layer<8, false>(A2, Bv, t2, nullptr); Bv = pack_B(t2);
    wmma_layer<8, false>(A3, Bv, t,  nullptr); Bv = pack_B(t);
    wmma_layer<8, false>(A4, Bv, t,  nullptr); Bv = pack_B(t);
    wmma_layer<8, true >(A5, Bv, t5, t2);      Bv = pack_B(t5);
    wmma_layer<8, false>(A6, Bv, t,  nullptr); Bv = pack_B(t);
    wmma_layer<8, true >(A7, Bv, t7, t5);      Bv = pack_B(t7);
    wmma_layer<8, false>(A8, Bv, t,  nullptr); Bv = pack_B(t);
    wmma_layer<8, true >(A9, Bv, t,  t7);      Bv = pack_B(t);
    wmma_layer<5, false>(A10, Bv, t, nullptr); Bv = pack_B(t);
    wmma_layer<3, false>(A11, Bv, t, nullptr); Bv = pack_B(t);

    // Layer 12: linear only (1 output feature), bias already in the GEMM.
    v8f c = {};
    v8f d = __builtin_amdgcn_wmma_f32_16x16x32_f16(
        false, A12, false, Bv, (short)0, c, false, false);
    if (lo) out[tile * 16 + m] = d[0];
  }
}

extern "C" void kernel_launch(void* const* d_in, const int* in_sizes, int n_in,
                              void* d_out, int out_size, void* d_ws,
                              size_t ws_size, hipStream_t stream) {
  (void)n_in; (void)out_size; (void)d_ws; (void)ws_size;
  const float* x = (const float*)d_in[0];
  // setup_inputs order: x, then W1,b1 ... W12,b12
  const float* W[12];
  const float* b[12];
  for (int i = 0; i < 12; ++i) {
    W[i] = (const float*)d_in[1 + 2 * i];
    b[i] = (const float*)d_in[2 + 2 * i];
  }
  float* out = (float*)d_out;

  const int rows   = in_sizes[0] / 8;  // x is [B, 8]
  const int ntiles = rows / 16;        // B = 2^22, divisible by 16

  dim3 grid(2048), block(256);  // 16384 waves, grid-stride over 262144 tiles
  mlp12_wmma_kernel<<<grid, block, 0, stream>>>(
      x, out, ntiles,
      W[0], b[0], W[1], b[1], W[2], b[2], W[3], b[3], W[4], b[4], W[5], b[5],
      W[6], b[6], W[7], b[7], W[8], b[8], W[9], b[9], W[10], b[10], W[11], b[11]);
}
